// SNSelfAttn_60498909331469
// MI455X (gfx1250) — matrix operand order repaired and checked by reading, not compiled, not run on hardware
//
#include <hip/hip_runtime.h>

#define DEVINL __device__ __forceinline__

typedef __attribute__((ext_vector_type(16))) __bf16 v16bf;
typedef __attribute__((ext_vector_type(8)))  __bf16 v8bf;
typedef __attribute__((ext_vector_type(8)))  float  v8f;

namespace {
constexpr int kB  = 8;
constexpr int kC  = 512;
constexpr int kL  = 4096;
constexpr int kCK = 64;        // C/8
constexpr int kCV = 256;       // C/2
constexpr int kM  = kL / 2;    // pooled key count = 2048

constexpr int kGPad = 40;      // 32 + 8 pad (LDS bank spread)
constexpr int kFPad = 72;      // 64 + 8 pad
}

// ---------- helpers ----------

// Native fptrunc -> hardware bf16 conversion (v_cvt_pk_bf16_f32).
DEVINL __bf16 tobf(float f) { return (__bf16)f; }
DEVINL unsigned short bfbits(float f) {
  union { __bf16 b; unsigned short s; } u; u.b = (__bf16)f; return u.s;
}

DEVINL v8f wmma_bf16(v16bf a, v16bf b, v8f c) {
  // D = A(16x32 bf16) * B(32x16 bf16) + C(16x16 f32)
  return __builtin_amdgcn_wmma_f32_16x16x32_bf16(
      /*neg_a=*/false, a, /*neg_b=*/false, b,
      /*c_mod=*/(short)0, c, /*reuse_a=*/false, /*reuse_b=*/false);
}

// 16 consecutive bf16 (two 16B loads; 16B alignment required).
DEVINL v16bf load_row16(const unsigned short* p) {
  v8bf lo = *(const v8bf*)p;
  v8bf hi = *(const v8bf*)(p + 8);
  v16bf r;
#pragma unroll
  for (int i = 0; i < 8; ++i) { r[i] = lo[i]; r[i + 8] = hi[i]; }
  return r;
}

// A fragment (16x32 bf16, ISA layout) from row-major bf16: p -> row start,
// elements 0..7 = K h*8..h*8+7 ; elements 8..15 = K 16+h*8..16+h*8+7
DEVINL v16bf load_a_bf16(const unsigned short* p, int h) {
  v8bf lo = *(const v8bf*)(p + h * 8);
  v8bf hi = *(const v8bf*)(p + 16 + h * 8);
  v16bf a;
#pragma unroll
  for (int i = 0; i < 8; ++i) { a[i] = lo[i]; a[i + 8] = hi[i]; }
  return a;
}

// A fragment from row-major fp32 (convert to bf16 in regs). p -> row start.
DEVINL v16bf load_a_f32(const float* p, int h) {
  const float4 q0 = *(const float4*)(p + h * 8);
  const float4 q1 = *(const float4*)(p + h * 8 + 4);
  const float4 q2 = *(const float4*)(p + 16 + h * 8);
  const float4 q3 = *(const float4*)(p + 16 + h * 8 + 4);
  v16bf a;
  a[0]=tobf(q0.x); a[1]=tobf(q0.y); a[2]=tobf(q0.z); a[3]=tobf(q0.w);
  a[4]=tobf(q1.x); a[5]=tobf(q1.y); a[6]=tobf(q1.z); a[7]=tobf(q1.w);
  a[8]=tobf(q2.x); a[9]=tobf(q2.y); a[10]=tobf(q2.z); a[11]=tobf(q2.w);
  a[12]=tobf(q3.x); a[13]=tobf(q3.y); a[14]=tobf(q3.z); a[15]=tobf(q3.w);
  return a;
}

// ---------- kernel 0: x[B,C,L] f32 -> xT[B,L,C] bf16 (tiled transpose) ----------
__global__ __launch_bounds__(256)
void cvt_transpose_kernel(const float* __restrict__ x,
                          unsigned short* __restrict__ xT) {
  __shared__ unsigned short tile[32][33];   // +1 pad: no bank conflicts
  const int tx = threadIdx.x;               // 0..31
  const int ty = threadIdx.y;               // 0..7
  const int l0 = blockIdx.x * 32;
  const int c0 = blockIdx.y * 32;
  const int b  = blockIdx.z;

  const float* xb = x + ((size_t)b * kC + c0) * kL + l0;
#pragma unroll
  for (int q = 0; q < 4; ++q) {
    const int ci = ty + q * 8;
    tile[ci][tx] = bfbits(xb[(size_t)ci * kL + tx]);   // coalesced 128B rows
  }
  __syncthreads();
  unsigned short* xo = xT + ((size_t)b * kL + l0) * kC + c0;
#pragma unroll
  for (int q = 0; q < 4; ++q) {
    const int li = ty + q * 8;
    xo[(size_t)li * kC + tx] = tile[tx][li];           // coalesced 64B rows
  }
}

// ---------- kernel 1: fused projections + maxpool2 ----------
// Y = W @ x[b]   (W: [R x 512] fp32, xT: [B,L,512] bf16)
//   rt 0..3  -> thetaT[B,L,64]  (bf16, no pool)
//   rt 4..7  -> phiT  [B,M,64]  (bf16, maxpool2 over l)
//   rt 8..23 -> g     [B,256,M] (bf16, maxpool2 over l)
__global__ __launch_bounds__(128)
void proj_kernel(const unsigned short* __restrict__ xT,
                 const float* __restrict__ Wt,
                 const float* __restrict__ Wp,
                 const float* __restrict__ Wg,
                 unsigned short* __restrict__ thetaT,
                 unsigned short* __restrict__ phiT,
                 unsigned short* __restrict__ gbf) {
  const int n    = threadIdx.x & 15;
  const int h    = threadIdx.x >> 4;
  const int ct   = blockIdx.x * 4 + threadIdx.y;   // 0..127, 32 cols each
  const int rt   = blockIdx.y;                     // 0..23
  const int b    = blockIdx.z;

  const float* W; int r0, kind;
  if (rt < 4)      { W = Wt; r0 = rt * 16;       kind = 0; }
  else if (rt < 8) { W = Wp; r0 = (rt - 4) * 16; kind = 1; }
  else             { W = Wg; r0 = (rt - 8) * 16; kind = 2; }

  const int l0 = ct * 32;
  const unsigned short* x0 = xT + ((size_t)b * kL + l0 + n) * kC;
  const unsigned short* x1 = xT + ((size_t)b * kL + l0 + 16 + n) * kC;
  const float* ap0 = W + (size_t)(r0 + n) * kC;

  v8f acc0 = {}, acc1 = {};
#pragma unroll
  for (int ks = 0; ks < kC / 32; ++ks) {
    const int kk = ks * 32;
    v16bf a  = load_a_f32(ap0 + kk, h);
    // B fragment: same k order as A (k = kk + h*8 + i, kk + 16 + h*8 + i)
    v16bf b0 = load_a_bf16(x0 + kk, h);
    v16bf b1 = load_a_bf16(x1 + kk, h);
    acc0 = wmma_bf16(a, b0, acc0);
    acc1 = wmma_bf16(a, b1, acc1);
  }

  if (kind == 0) {                              // thetaT[l][k]
    unsigned short* tp = thetaT + (size_t)b * kL * kCK;
#pragma unroll
    for (int r = 0; r < 8; ++r) {
      const int k = r0 + r + 8 * h;
      tp[(size_t)(l0 + n) * kCK + k]      = bfbits(acc0[r]);
      tp[(size_t)(l0 + 16 + n) * kCK + k] = bfbits(acc1[r]);
    }
  } else {                                      // fused maxpool2 over l pairs
    const int mo = l0 >> 1;                     // 16 pooled cols per tile
#pragma unroll
    for (int r = 0; r < 8; ++r) {
      float p0 = fmaxf(acc0[r], __shfl_xor(acc0[r], 1, 16));
      float p1 = fmaxf(acc1[r], __shfl_xor(acc1[r], 1, 16));
      if ((n & 1) == 0) {
        const int j = n >> 1;
        if (kind == 1) {                        // phiT[m][k]
          const int k = r0 + r + 8 * h;
          unsigned short* pp = phiT + (size_t)b * kM * kCK;
          pp[(size_t)(mo + j) * kCK + k]     = bfbits(p0);
          pp[(size_t)(mo + 8 + j) * kCK + k] = bfbits(p1);
        } else {                                // g[v][m]
          const int v = r0 + r + 8 * h;
          unsigned short* gp = gbf + (size_t)b * kCV * kM;
          gp[(size_t)v * kM + mo + j]     = bfbits(p0);
          gp[(size_t)v * kM + mo + 8 + j] = bfbits(p1);
        }
      }
    }
  }
}

// ---------- kernel 2: flash attention over pooled keys ----------
// 8 wave32 tiles / block share double-buffered LDS stages of the g and phi
// chunks (8x L2-traffic reduction); next chunk's global loads are issued
// before the compute phase so they overlap the WMMA work.
__global__ __launch_bounds__(256)
void attn_kernel(const unsigned short* __restrict__ thetaT,
                 const unsigned short* __restrict__ phiT,
                 const unsigned short* __restrict__ gbf,
                 unsigned short* __restrict__ oT) {
  __shared__ __align__(32) unsigned short gbuf[2][kCV][kGPad];  // 40 KB
  __shared__ __align__(32) unsigned short fbuf[2][32][kFPad];   //  9 KB
  __shared__ __align__(32) unsigned short pbuf[8][16 * 32];     //  8 KB

  const int n    = threadIdx.x & 15;
  const int h    = threadIdx.x >> 4;
  const int wave = threadIdx.y;                     // 0..7
  const int tid  = wave * 32 + threadIdx.x;         // 0..255
  const int lt   = blockIdx.x * 8 + wave;           // 0..255
  const int b    = blockIdx.z;
  const int l0   = lt * 16;

  const unsigned short* tp = thetaT + ((size_t)b * kL + l0 + n) * kCK;
  const unsigned short* pp = phiT + (size_t)b * kM * kCK;
  const unsigned short* gp = gbf + (size_t)b * kCV * kM;

  // per-thread staging sources: one g row (v = tid) and one phi 16B slice
  const unsigned short* grow = gp + (size_t)tid * kM;
  const unsigned short* frow = pp + (size_t)(tid >> 3) * kCK + (tid & 7) * 8;

  // query A fragments, reused for every chunk (K = 64 -> two 32-wide steps)
  const v16bf a0 = load_a_bf16(tp, h);
  const v16bf a1 = load_a_bf16(tp + 32, h);

  float mr[8], lr[8];
#pragma unroll
  for (int r = 0; r < 8; ++r) { mr[r] = -__builtin_inff(); lr[r] = 0.0f; }
  v8f zero = {};
  v8f o[16];
#pragma unroll
  for (int t = 0; t < 16; ++t) o[t] = zero;

  unsigned short* pb = pbuf[wave];

  // ---- prefill buffer 0 with chunk m0 = 0 ----
  {
    uint4 q0 = *(const uint4*)(grow);
    uint4 q1 = *(const uint4*)(grow + 8);
    uint4 q2 = *(const uint4*)(grow + 16);
    uint4 q3 = *(const uint4*)(grow + 24);
    uint4 f0 = *(const uint4*)(frow);
    *(uint4*)&gbuf[0][tid][0]  = q0;
    *(uint4*)&gbuf[0][tid][8]  = q1;
    *(uint4*)&gbuf[0][tid][16] = q2;
    *(uint4*)&gbuf[0][tid][24] = q3;
    *(uint4*)&fbuf[0][tid >> 3][(tid & 7) * 8] = f0;
  }
  __syncthreads();

  int cur = 0;
  for (int m0 = 0; m0 < kM; m0 += 32) {
    const int nxt = cur ^ 1;
    const bool have_next = (m0 + 32) < kM;

    // ---- issue next chunk's global loads (overlap with compute) ----
    uint4 q0, q1, q2, q3, f0;
    if (have_next) {
      const unsigned short* gr = grow + m0 + 32;
      q0 = *(const uint4*)(gr);
      q1 = *(const uint4*)(gr + 8);
      q2 = *(const uint4*)(gr + 16);
      q3 = *(const uint4*)(gr + 24);
      f0 = *(const uint4*)(frow + (size_t)(m0 + 32) * kCK);
      __builtin_prefetch(gr + 32, 0, 0);   // chunk after next
    }

    // ---- scores S[16 x 32] from LDS phi stage ----
    v8f s0 = zero, s1 = zero;
    s0 = wmma_bf16(a0, load_a_bf16(&fbuf[cur][n][0], h), s0);
    s0 = wmma_bf16(a1, load_a_bf16(&fbuf[cur][n][32], h), s0);
    s1 = wmma_bf16(a0, load_a_bf16(&fbuf[cur][16 + n][0], h), s1);
    s1 = wmma_bf16(a1, load_a_bf16(&fbuf[cur][16 + n][32], h), s1);

    // ---- online softmax (row = r + 8h lives in one 16-lane half) ----
    float sc[8];
#pragma unroll
    for (int r = 0; r < 8; ++r) {
      float cm = fmaxf(s0[r], s1[r]);
      cm = fmaxf(cm, __shfl_xor(cm, 8, 16));
      cm = fmaxf(cm, __shfl_xor(cm, 4, 16));
      cm = fmaxf(cm, __shfl_xor(cm, 2, 16));
      cm = fmaxf(cm, __shfl_xor(cm, 1, 16));
      const float mnew = fmaxf(mr[r], cm);
      sc[r] = __expf(mr[r] - mnew);
      const float e0 = __expf(s0[r] - mnew);
      const float e1 = __expf(s1[r] - mnew);
      s0[r] = e0; s1[r] = e1;
      float rs = e0 + e1;
      rs += __shfl_xor(rs, 8, 16);
      rs += __shfl_xor(rs, 4, 16);
      rs += __shfl_xor(rs, 2, 16);
      rs += __shfl_xor(rs, 1, 16);
      lr[r] = lr[r] * sc[r] + rs;
      mr[r] = mnew;
    }
#pragma unroll
    for (int t = 0; t < 16; ++t)
#pragma unroll
      for (int r = 0; r < 8; ++r) o[t][r] *= sc[r];

    // ---- C-layout -> A-fragment relayout of P through per-wave LDS ----
#pragma unroll
    for (int r = 0; r < 8; ++r) {
      const int row = r + 8 * h;
      pb[row * 32 + n]      = bfbits(s0[r]);
      pb[row * 32 + 16 + n] = bfbits(s1[r]);
    }
    asm volatile("s_wait_dscnt 0" ::: "memory");
    const v16bf pa = load_a_bf16(pb + n * 32, h);

    // ---- O[16x256] += P[16x32] @ gT[32x256] from LDS g stage ----
#pragma unroll
    for (int t = 0; t < 16; ++t) {
      const v16bf gfrag = load_a_bf16(&gbuf[cur][t * 16 + n][0], h);
      o[t] = wmma_bf16(pa, gfrag, o[t]);
    }

    // ---- commit next chunk to LDS and rotate buffers ----
    if (have_next) {
      *(uint4*)&gbuf[nxt][tid][0]  = q0;
      *(uint4*)&gbuf[nxt][tid][8]  = q1;
      *(uint4*)&gbuf[nxt][tid][16] = q2;
      *(uint4*)&gbuf[nxt][tid][24] = q3;
      *(uint4*)&fbuf[nxt][tid >> 3][(tid & 7) * 8] = f0;
    }
    __syncthreads();
    cur = nxt;
  }

  // ---- normalize + store o_inT[B, L, 256] bf16 ----
  float inv[8];
#pragma unroll
  for (int r = 0; r < 8; ++r) inv[r] = 1.0f / lr[r];
  unsigned short* op = oT + (size_t)b * kL * kCV;
#pragma unroll
  for (int t = 0; t < 16; ++t)
#pragma unroll
    for (int r = 0; r < 8; ++r) {
      const int l = l0 + r + 8 * h;
      op[(size_t)l * kCV + t * 16 + n] = bfbits(o[t][r] * inv[r]);
    }
}

// ---------- kernel 3: output projection + residual ----------
// out = gamma * (W_o @ o_in) + x
__global__ __launch_bounds__(128)
void out_kernel(const float* __restrict__ Wo,
                const unsigned short* __restrict__ oT,
                const float* __restrict__ x,
                const float* __restrict__ gamma,
                float* __restrict__ out) {
  const int n  = threadIdx.x & 15;
  const int h  = threadIdx.x >> 4;
  const int ct = blockIdx.x * 4 + threadIdx.y;   // 0..127
  const int rt = blockIdx.y;                     // 0..31
  const int b  = blockIdx.z;
  const int o0 = rt * 16, l0 = ct * 32;

  const float* ap0 = Wo + (size_t)(o0 + n) * kCV;
  const unsigned short* bp = oT + (size_t)b * kL * kCV;

  v8f acc0 = {}, acc1 = {};
#pragma unroll
  for (int ks = 0; ks < kCV / 32; ++ks) {
    const int kk = ks * 32;
    v16bf a = load_a_f32(ap0 + kk, h);
    v16bf b0 = load_a_bf16(bp + (size_t)(l0 + n) * kCV + kk, h);
    v16bf b1 = load_a_bf16(bp + (size_t)(l0 + 16 + n) * kCV + kk, h);
    acc0 = wmma_bf16(a, b0, acc0);
    acc1 = wmma_bf16(a, b1, acc1);
  }

  const float gm = *gamma;
#pragma unroll
  for (int r = 0; r < 8; ++r) {
    const int oc = o0 + r + 8 * h;
    const size_t base = ((size_t)b * kC + oc) * kL;
    const int la = l0 + n, lb = l0 + 16 + n;
    out[base + la] = gm * acc0[r] + x[base + la];
    out[base + lb] = gm * acc1[r] + x[base + lb];
  }
}

// ---------- launcher ----------
extern "C" void kernel_launch(void* const* d_in, const int* in_sizes, int n_in,
                              void* d_out, int out_size, void* d_ws, size_t ws_size,
                              hipStream_t stream) {
  (void)in_sizes; (void)n_in; (void)out_size; (void)ws_size;
  const float* x     = (const float*)d_in[0];
  const float* Wt    = (const float*)d_in[1];
  const float* Wp    = (const float*)d_in[2];
  const float* Wg    = (const float*)d_in[3];
  const float* Wo    = (const float*)d_in[4];
  const float* gamma = (const float*)d_in[5];
  float* out = (float*)d_out;

  unsigned short* xT     = (unsigned short*)d_ws;                 // [B,L,512]
  unsigned short* thetaT = xT + (size_t)kB * kL * kC;             // [B,L,64]
  unsigned short* phiT   = thetaT + (size_t)kB * kL * kCK;        // [B,M,64]
  unsigned short* gbf    = phiT + (size_t)kB * kM * kCK;          // [B,256,M]
  unsigned short* oT     = gbf + (size_t)kB * kCV * kM;           // [B,L,256]

  cvt_transpose_kernel<<<dim3(kL / 32, kC / 32, kB), dim3(32, 8, 1), 0, stream>>>(
      x, xT);

  const dim3 blk4(32, 4, 1);
  proj_kernel<<<dim3(kL / 32 / 4, 24, kB), blk4, 0, stream>>>(
      xT, Wt, Wp, Wg, thetaT, phiT, gbf);
  attn_kernel<<<dim3(kL / 16 / 8, 1, kB), dim3(32, 8, 1), 0, stream>>>(
      thetaT, phiT, gbf, oT);
  out_kernel<<<dim3(kL / 32 / 4, kC / 16, kB), blk4, 0, stream>>>(
      Wo, oT, x, gamma, out);
}
